// Autoencoder_84851373900034
// MI455X (gfx1250) — matrix operand order repaired and checked by reading, not compiled, not run on hardware
//
#include <hip/hip_runtime.h>

// ---------- vector types ----------
typedef __attribute__((ext_vector_type(16))) __bf16   v16bf;
typedef __attribute__((ext_vector_type(8)))  float    v8f;
typedef __attribute__((ext_vector_type(4)))  float    v4f;
typedef __attribute__((ext_vector_type(4)))  unsigned int v4u;

// ---------- network shape ----------
constexpr int DIMS[13] = {512, 256, 128, 64, 32, 16, 5, 16, 32, 64, 128, 256, 512};
constexpr int BATCH = 131072;

constexpr int ktiles(int in)  { return (in  + 31) / 32; }
constexpr int ntiles(int out) { return (out + 15) / 16; }

// fragment element offset (in bf16 elements) of layer l in the workspace
constexpr long foff(int l) {
    long a = 0;
    for (int i = 0; i < l; ++i)
        a += (long)ktiles(DIMS[i]) * ntiles(DIMS[i + 1]) * 512;
    return a;
}

struct LayerPtrs { const float* w; const float* b; };
struct Params    { LayerPtrs l[12]; };

// LDS activation buffer strides (bf16 elements, +8 pad to spread banks)
constexpr int S0 = 512 + 8;   // wide ping buffer
constexpr int S1 = 256 + 8;   // narrow pong buffer
constexpr int WAVES = 4;      // waves per workgroup

// =====================================================================
// Prep: fp32 weights -> bf16 WMMA B-fragments in workspace.
// B fragment for v_wmma_f32_16x16x32_bf16 (32x16, K x N):
//   lane n     (n<16): column N=n,  K = kbase+{0..7, 16..23}
//   lane n+16        : column N=n,  K = kbase+{8..15, 24..31}
// Stored so each lane's 32 bytes are contiguous: frag[fragIdx][lane][16].
// Out-of-range K/N (5- and 16-dim layers) are zero-filled.
// =====================================================================
__global__ void prep_kernel(Params p, unsigned short* frags_raw) {
    __bf16* frags = (__bf16*)frags_raw;
    long idx = (long)blockIdx.x * blockDim.x + threadIdx.x;
    constexpr long TOTAL = foff(12);
    if (idx >= TOTAL) return;

    // locate layer
    int l = 0; long base = 0;
#pragma unroll
    for (int i = 0; i < 12; ++i) {
        constexpr_wrap: ;
        long sz = (long)ktiles(DIMS[i]) * ntiles(DIMS[i + 1]) * 512;
        if (idx >= base && idx < base + sz) { l = i; }
        if (idx >= base + sz) base += sz;
    }
    int in  = DIMS[l];
    int out = DIMS[l + 1];
    int KT  = ktiles(in);

    long local = idx - base;
    int fragi = (int)(local >> 9);      // fragment index (nt*KT + kt)
    int j     = (int)(local & 511);     // element within fragment
    int lane  = j >> 4;
    int e     = j & 15;
    int nt = fragi / KT;
    int kt = fragi % KT;
    int col = nt * 16 + (lane & 15);
    int k   = kt * 32 + (lane >> 4) * 8 + (e & 7) + (e >> 3) * 16;

    float v = 0.0f;
    if (k < in && col < out) v = p.l[l].w[(long)k * out + col];
    frags[idx] = (__bf16)v;
}

// =====================================================================
// One fused layer for a single wave over its private 16 rows.
// A fragment (16x32 bf16) per documented layout:
//   lane m (m<16): row M=m, K = kbase+{0..7, 16..23}   -> two ds_load_b128
//   lane m+16    : row M=m, K = kbase+{8..15, 24..31}
// =====================================================================
template <int IN, int OUT, bool LAST>
__device__ __forceinline__ void run_layer(const __bf16* __restrict__ frag,
                                          const float* __restrict__ bias,
                                          const __bf16* __restrict__ inbuf, int istride,
                                          __bf16* __restrict__ outbuf, int ostride,
                                          float* __restrict__ gout, int lane) {
    constexpr int KT = ktiles(IN);
    constexpr int NT = ntiles(OUT);

    const int m  = lane & 15;
    const int hi = lane >> 4;                       // 0 or 1
    const __bf16* arow  = inbuf + m * istride + hi * 8;
    const __bf16* bbase = frag + lane * 16;

    for (int nt = 0; nt < NT; nt += 2) {
        const bool two = (nt + 1) < NT;
        v8f acc0 = {};
        v8f acc1 = {};
        const __bf16* bp0 = bbase + (long)nt * KT * 512;
        const __bf16* bp1 = bp0 + (long)KT * 512;

#pragma unroll
        for (int kt = 0; kt < KT; ++kt) {
            union { v16bf v; v4u q[2]; } a, b0, b1;
            const __bf16* ap = arow + kt * 32;
            a.q[0] = *(const v4u*)(ap);             // K 0..7   (or 8..15)
            a.q[1] = *(const v4u*)(ap + 16);        // K 16..23 (or 24..31)
            b0.q[0] = *(const v4u*)(bp0 + kt * 512);
            b0.q[1] = *(const v4u*)(bp0 + kt * 512 + 8);
            acc0 = __builtin_amdgcn_wmma_f32_16x16x32_bf16(
                       false, a.v, false, b0.v, (short)0, acc0, false, false);
            if (two) {
                b1.q[0] = *(const v4u*)(bp1 + kt * 512);
                b1.q[1] = *(const v4u*)(bp1 + kt * 512 + 8);
                acc1 = __builtin_amdgcn_wmma_f32_16x16x32_bf16(
                           false, a.v, false, b1.v, (short)0, acc1, false, false);
            }
        }

        // epilogue: bias + ReLU, D layout: VGPR r -> M = r + hi*8, N = lane&15
        const int m0 = hi * 8;
        {
            const int col = nt * 16 + (lane & 15);
            const float bv = (col < OUT) ? bias[col] : 0.0f;
#pragma unroll
            for (int r = 0; r < 8; ++r) {
                float v = acc0[r] + bv;
                v = v > 0.0f ? v : 0.0f;
                if (LAST) gout[(m0 + r) * 512 + col] = v;
                else      outbuf[(m0 + r) * ostride + col] = (__bf16)v;
            }
        }
        if (two) {
            const int col = (nt + 1) * 16 + (lane & 15);
            const float bv = (col < OUT) ? bias[col] : 0.0f;
#pragma unroll
            for (int r = 0; r < 8; ++r) {
                float v = acc1[r] + bv;
                v = v > 0.0f ? v : 0.0f;
                if (LAST) gout[(m0 + r) * 512 + col] = v;
                else      outbuf[(m0 + r) * ostride + col] = (__bf16)v;
            }
        }
    }

    // zero-pad K columns [16,32) so the next layer always reads K%32==0
    if (!LAST && OUT < 32) {
        v4u z = {0u, 0u, 0u, 0u};
        *(v4u*)(outbuf + m * ostride + 16 + hi * 8) = z;
    }
}

// =====================================================================
// Fused autoencoder: 4 independent wave32s / block, 16 rows each.
// Activations ping-pong between two per-wave LDS buffers; no barriers.
// =====================================================================
__global__ __launch_bounds__(WAVES * 32) void ae_kernel(const float* __restrict__ x,
                                                        float* __restrict__ out,
                                                        const unsigned short* __restrict__ frags_raw,
                                                        Params p) {
    extern __shared__ char smem[];
    const __bf16* frags = (const __bf16*)frags_raw;
    const int wid  = threadIdx.x >> 5;
    const int lane = threadIdx.x & 31;

    __bf16* buf0 = (__bf16*)smem + (long)wid * 16 * S0;
    __bf16* buf1 = (__bf16*)(smem + (size_t)WAVES * 16 * S0 * sizeof(__bf16))
                   + (long)wid * 16 * S1;

    const long rowbase = ((long)blockIdx.x * WAVES + wid) * 16;
    const float* xr = x + rowbase * 512;
    float* gout = out + rowbase * 512;

    // stage this wave's 16x512 input tile into LDS as bf16 (coalesced b128 loads)
#pragma unroll 4
    for (int i = lane * 4; i < 16 * 512; i += 32 * 4) {
        int r = i >> 9;
        int c = i & 511;
        v4f xv = *(const v4f*)(xr + (long)r * 512 + c);
        union { __bf16 h[4]; unsigned long long u; } t;
        t.h[0] = (__bf16)xv[0]; t.h[1] = (__bf16)xv[1];
        t.h[2] = (__bf16)xv[2]; t.h[3] = (__bf16)xv[3];
        *(unsigned long long*)(buf0 + r * S0 + c) = t.u;
    }

    run_layer<512, 256, false>(frags + foff(0),  p.l[0].b,  buf0, S0, buf1, S1, nullptr, lane);
    run_layer<256, 128, false>(frags + foff(1),  p.l[1].b,  buf1, S1, buf0, S0, nullptr, lane);
    run_layer<128,  64, false>(frags + foff(2),  p.l[2].b,  buf0, S0, buf1, S1, nullptr, lane);
    run_layer< 64,  32, false>(frags + foff(3),  p.l[3].b,  buf1, S1, buf0, S0, nullptr, lane);
    run_layer< 32,  16, false>(frags + foff(4),  p.l[4].b,  buf0, S0, buf1, S1, nullptr, lane);
    run_layer< 16,   5, false>(frags + foff(5),  p.l[5].b,  buf1, S1, buf0, S0, nullptr, lane);
    run_layer<  5,  16, false>(frags + foff(6),  p.l[6].b,  buf0, S0, buf1, S1, nullptr, lane);
    run_layer< 16,  32, false>(frags + foff(7),  p.l[7].b,  buf1, S1, buf0, S0, nullptr, lane);
    run_layer< 32,  64, false>(frags + foff(8),  p.l[8].b,  buf0, S0, buf1, S1, nullptr, lane);
    run_layer< 64, 128, false>(frags + foff(9),  p.l[9].b,  buf1, S1, buf0, S0, nullptr, lane);
    run_layer<128, 256, false>(frags + foff(10), p.l[10].b, buf0, S0, buf1, S1, nullptr, lane);
    run_layer<256, 512, true >(frags + foff(11), p.l[11].b, buf1, S1, nullptr, 0, gout, lane);
}

// =====================================================================
extern "C" void kernel_launch(void* const* d_in, const int* in_sizes, int n_in,
                              void* d_out, int out_size, void* d_ws, size_t ws_size,
                              hipStream_t stream) {
    (void)in_sizes; (void)n_in; (void)out_size; (void)ws_size;

    const float* x = (const float*)d_in[0];
    Params p;
    for (int l = 0; l < 12; ++l) {
        p.l[l].w = (const float*)d_in[1 + 2 * l];
        p.l[l].b = (const float*)d_in[2 + 2 * l];
    }
    unsigned short* frags = (unsigned short*)d_ws;   // ~701 KB of bf16 fragments

    constexpr long TOTAL = foff(12);                 // 350,720 elements
    prep_kernel<<<(int)((TOTAL + 255) / 256), 256, 0, stream>>>(p, frags);

    const size_t smem_bytes =
        (size_t)WAVES * 16 * (S0 + S1) * sizeof(unsigned short);   // 100,352 B
    const int blocks = BATCH / (WAVES * 16);                       // 2048
    ae_kernel<<<blocks, WAVES * 32, smem_bytes, stream>>>(x, (float*)d_out, frags, p);
}
